// HybridModelFinal_49761491091476
// MI455X (gfx1250) — matrix operand order repaired and checked by reading, not compiled, not run on hardware
//
#include <hip/hip_runtime.h>
#include <hip/hip_bf16.h>

typedef __attribute__((ext_vector_type(16))) _Float16 v16h;
typedef __attribute__((ext_vector_type(8)))  float    v8f;

// ---------------------------------------------------------------------------
// Kernel 1: fused conv1 -> relu -> pool -> conv2 -> relu -> pool, per sample.
// Both convs are implicit GEMMs on v_wmma_f32_16x16x32_f16.
// Image is staged global->LDS with async-to-LDS (ASYNCcnt) B128 transfers.
// Padded-K lanes of the A fragments are left as garbage: the matching
// B-fragment rows are zero, so the WMMA products vanish.
// ---------------------------------------------------------------------------
__global__ __launch_bounds__(128)
void conv_fused_kernel(const float* __restrict__ x,
                       const float* __restrict__ w1, const float* __restrict__ b1,
                       const float* __restrict__ w2, const float* __restrict__ b2,
                       _Float16* __restrict__ h2)
{
    __shared__ float    img32[784];           // raw f32 image (async staged)
    __shared__ _Float16 img[784 + 64];        // f16 image (+pad: padded-k reads)
    __shared__ _Float16 c1o[576 * 16];        // conv1 out (24x24, ch stride 16)
    __shared__ _Float16 p1[12 * 12 * 8];      // pooled1 [y*12+x][8] (c stride 8)
    __shared__ _Float16 w2s[16 * 160];        // conv2 weights [n][k], K pad 160 (zeros)
    __shared__ _Float16 c2o[64 * 16];         // conv2 out (8x8, 16ch)

    const int b     = blockIdx.x;
    const int tid   = threadIdx.x;
    const int lane  = tid & 31;
    const int wave  = tid >> 5;               // 0..3
    const int lhalf = (lane >= 16) ? 1 : 0;
    const int ncol  = lane & 15;

    // ---- async stage of the raw image: 196 x B128 global->LDS transfers ----
    {
        const float* xb = x + (size_t)b * 784;
        unsigned lbase = (unsigned)(size_t)(&img32[0]);   // LDS byte offset
        for (int p = tid; p < 196; p += 128) {
            unsigned loff = lbase + (unsigned)p * 16u;
            const float* g = xb + p * 4;
            asm volatile("global_load_async_to_lds_b128 %0, %1, off"
                         :: "v"(loff), "v"(g) : "memory");
        }
        asm volatile("s_wait_asynccnt 0x0" ::: "memory");
    }
    // stage conv2 weights (f32 -> f16, zero K-pad) after the async wait
    for (int p = tid; p < 16 * 160; p += 128) {
        int n = p / 160, k = p % 160;
        w2s[p] = (k < 150) ? (_Float16)w2[n * 150 + k] : (_Float16)0.f;
    }
    __syncthreads();
    for (int p = tid; p < 784; p += 128) img[p] = (_Float16)img32[p];
    __syncthreads();

    // ---- conv1: (576 x 25) * (25 x 6), K padded to 32, N padded to 16 ----
    // B fragment (reused for all 9 M tiles): MUST be zero at k>=25 / n>=6.
    v16h bfrag1;
    {
        int kb = lhalf * 8;
        int nc = (ncol < 6) ? ncol : 0;
        #pragma unroll
        for (int h = 0; h < 16; ++h) {
            int k  = kb + ((h < 8) ? h : (h + 8));
            float v = w1[nc * 25 + ((k < 25) ? k : 0)];
            bfrag1[h] = (ncol < 6 && k < 25) ? (_Float16)v : (_Float16)0.f;
        }
    }
    const float bias1 = b1[(ncol < 6) ? ncol : 0];   // junk channels never read

    for (int t = 0; t < 9; ++t) {
        int tile = wave * 9 + t;
        int m0 = tile * 16;
        int m  = m0 + ncol;                   // A row for this lane
        int y  = m / 24, xq = m % 24;
        int base = y * 28 + xq;
        int kb = lhalf * 8;
        v16h afrag;
        #pragma unroll
        for (int h = 0; h < 16; ++h) {
            int k  = kb + ((h < 8) ? h : (h + 8));   // K = kb+{0..7,16..23}
            int ki = k / 5, kj = k % 5;
            afrag[h] = img[base + ki * 28 + kj];     // k>=25: garbage * 0
        }
        v8f acc = {};
        acc = __builtin_amdgcn_wmma_f32_16x16x32_f16(false, afrag, false, bfrag1,
                                                     (short)0, acc, false, false);
        #pragma unroll
        for (int r = 0; r < 8; ++r) {
            int mr = m0 + r + lhalf * 8;
            float v = acc[r] + bias1;
            v = v > 0.f ? v : 0.f;
            c1o[mr * 16 + ncol] = (_Float16)v;       // unconditional, stride 16
        }
    }
    __syncthreads();

    // ---- pool1: 24x24x6 -> 12x12 x (stride 8); shifts/masks only + /12 ----
    for (int o = tid; o < 1152; o += 128) {
        int c = o & 7, sp = o >> 3, px = sp % 12, py = sp / 12;
        int base = ((2 * py) * 24 + 2 * px) * 16 + c;
        float v0 = (float)c1o[base],       v1 = (float)c1o[base + 16];
        float v2 = (float)c1o[base + 384], v3 = (float)c1o[base + 400];
        p1[o] = (_Float16)fmaxf(fmaxf(v0, v1), fmaxf(v2, v3));
    }
    __syncthreads();

    // ---- conv2: (64 x 150) * (150 x 16), K padded to 160 (B rows zero) ----
    {
        int m0 = wave * 16;                   // 4 waves -> 4 M tiles
        int m  = m0 + ncol;
        int y  = m / 8, xx = m % 8;
        int base = (y * 12 + xx) * 8;
        v8f acc = {};
        #pragma unroll
        for (int s = 0; s < 5; ++s) {
            v16h afrag;
            int kb = s * 32 + lhalf * 16;
            #pragma unroll
            for (int h = 0; h < 16; ++h) {
                int k  = kb + h;
                int c  = k / 25, rr = (k % 25) / 5, cc = k % 5;
                afrag[h] = p1[base + (rr * 12 + cc) * 8 + c];  // k>=150: *0
            }
            v16h bfrag;
            const _Float16* pb = &w2s[ncol * 160 + s * 32 + lhalf * 16];
            __builtin_memcpy(&bfrag, pb, 32);        // 2 x ds_load_b128
            acc = __builtin_amdgcn_wmma_f32_16x16x32_f16(false, afrag, false, bfrag,
                                                         (short)0, acc, false, false);
        }
        #pragma unroll
        for (int r = 0; r < 8; ++r) {
            int mr = m0 + r + lhalf * 8;
            float v = acc[r] + b2[ncol];
            v = v > 0.f ? v : 0.f;
            c2o[mr * 16 + ncol] = (_Float16)v;
        }
    }
    __syncthreads();

    // ---- pool2 + flatten (ch*16 + y*4 + x) -> h2[b][256] ----
    _Float16* outp = h2 + (size_t)b * 256;
    for (int o = tid; o < 256; o += 128) {
        int ch = o >> 4, rem = o & 15, yy = rem >> 2, xx = rem & 3;
        int base = ((2 * yy) * 8 + 2 * xx) * 16 + ch;
        float v0 = (float)c2o[base],       v1 = (float)c2o[base + 16];
        float v2 = (float)c2o[base + 128], v3 = (float)c2o[base + 144];
        outp[o] = (_Float16)fmaxf(fmaxf(v0, v1), fmaxf(v2, v3));
    }
}

// ---------------------------------------------------------------------------
// Kernel 2: FC1  h3 = relu(h2 @ W^T + b),  M=B, K=256, N=120 (pad 128), WMMA.
// 8 waves/block: wave = N tile; each wave does 4 M tiles; K loop of 8 steps.
// Columns n>=120 produce garbage that is never read downstream.
// ---------------------------------------------------------------------------
__global__ __launch_bounds__(256)
void fc1_kernel(const _Float16* __restrict__ h2,
                const float* __restrict__ w, const float* __restrict__ bias,
                _Float16* __restrict__ h3)
{
    const int tid    = threadIdx.x;
    const int lane   = tid & 31;
    const int wave   = tid >> 5;              // N tile 0..7
    const int lhalf  = (lane >= 16) ? 1 : 0;
    const int ncol   = lane & 15;
    const int m0     = blockIdx.x * 64;
    const int n      = wave * 16 + ncol;      // output column 0..127
    const int nc     = (n < 120) ? n : 0;     // clamp for safe loads only
    const float* wn  = w + (size_t)nc * 256;

    v8f acc[4] = {};
    for (int s = 0; s < 8; ++s) {
        // B fragment: row n of fc1_w (K contiguous), f32->f16 on the fly.
        v16h bfrag;
        const float* pw = wn + s * 32 + lhalf * 16;
        #pragma unroll
        for (int h = 0; h < 16; ++h) bfrag[h] = (_Float16)pw[h];
        #pragma unroll
        for (int mt = 0; mt < 4; ++mt) {
            int m = m0 + mt * 16 + ncol;
            const _Float16* pa = h2 + (size_t)m * 256 + s * 32 + lhalf * 16;
            v16h afrag;
            __builtin_memcpy(&afrag, pa, 16);                       // halves 0..7
            __builtin_memcpy((char*)&afrag + 16, pa + 8, 16);       // halves 8..15
            acc[mt] = __builtin_amdgcn_wmma_f32_16x16x32_f16(false, afrag, false, bfrag,
                                                             (short)0, acc[mt], false, false);
        }
    }
    const float bn = bias[nc];
    #pragma unroll
    for (int mt = 0; mt < 4; ++mt) {
        #pragma unroll
        for (int r = 0; r < 8; ++r) {
            int m = m0 + mt * 16 + r + lhalf * 8;
            float v = acc[mt][r] + bn;
            v = v > 0.f ? v : 0.f;
            h3[(size_t)m * 128 + n] = (_Float16)v;
        }
    }
}

// ---------------------------------------------------------------------------
// Kernel 3: FC2 + sigmoid -> 4-qubit statevector circuit -> classifier.
// One thread per sample; 16 complex amplitudes live in registers.
// Qubit w maps to bit (3 - w) of the flat state index.
// ---------------------------------------------------------------------------
__device__ __forceinline__ float2 cmul(float2 a, float2 b) {
    return make_float2(a.x * b.x - a.y * b.y, a.x * b.y + a.y * b.x);
}
__device__ __forceinline__ float2 cadd(float2 a, float2 b) {
    return make_float2(a.x + b.x, a.y + b.y);
}

__global__ __launch_bounds__(256)
void quantum_kernel(const _Float16* __restrict__ h3,
                    const float* __restrict__ fc2w, const float* __restrict__ fc2b,
                    const float* __restrict__ qw,
                    const float* __restrict__ cw, const float* __restrict__ cb,
                    float* __restrict__ out, int nbatch)
{
    int b = blockIdx.x * blockDim.x + threadIdx.x;
    if (b >= nbatch) return;
    const _Float16* h = h3 + (size_t)b * 128;

    float ang[4];
    #pragma unroll
    for (int j = 0; j < 4; ++j) {
        float a = fc2b[j];
        for (int k = 0; k < 120; ++k) a += fc2w[j * 120 + k] * (float)h[k];
        ang[j] = 3.14159265358979f / (1.f + expf(-a));
    }

    float2 st[16];
    #pragma unroll
    for (int i = 0; i < 16; ++i) st[i] = make_float2(0.f, 0.f);
    st[0].x = 1.f;

    for (int l = 0; l < 3; ++l) {
        // Rot(phi, theta, omega) on each qubit
        for (int w = 0; w < 4; ++w) {
            const float* p = qw + (l * 4 + w) * 3;
            float phi = p[0], th = p[1], om = p[2];
            float sa, ca, sb, cbb, stt, ct;
            sincosf(0.5f * (phi + om), &sa, &ca);
            sincosf(0.5f * (phi - om), &sb, &cbb);
            sincosf(0.5f * th, &stt, &ct);
            float2 m00 = make_float2( ca * ct, -sa * ct);
            float2 m01 = make_float2(-cbb * stt, -sb * stt);
            float2 m10 = make_float2( cbb * stt, -sb * stt);
            float2 m11 = make_float2( ca * ct,  sa * ct);
            int bit = 3 - w, msk = 1 << bit;
            #pragma unroll
            for (int p8 = 0; p8 < 8; ++p8) {
                int low = p8 & (msk - 1);
                int i   = ((p8 ^ low) << 1) | low;   // insert 0 at `bit`
                int j   = i | msk;
                float2 si = st[i], sj = st[j];
                st[i] = cadd(cmul(m00, si), cmul(m01, sj));
                st[j] = cadd(cmul(m10, si), cmul(m11, sj));
            }
        }
        // CNOT ring (w -> (w+1)%4)
        for (int w = 0; w < 4; ++w) {
            int cm = 1 << (3 - w), tm = 1 << (3 - ((w + 1) & 3));
            #pragma unroll
            for (int i = 0; i < 16; ++i) {
                if ((i & cm) && !(i & tm)) {
                    int j = i | tm;
                    float2 tmp = st[i]; st[i] = st[j]; st[j] = tmp;
                }
            }
        }
        // RX(angles[w]) on each qubit
        for (int w = 0; w < 4; ++w) {
            float s2, c2;
            sincosf(0.5f * ang[w], &s2, &c2);
            float2 m00 = make_float2(c2, 0.f), m01 = make_float2(0.f, -s2);
            int bit = 3 - w, msk = 1 << bit;
            #pragma unroll
            for (int p8 = 0; p8 < 8; ++p8) {
                int low = p8 & (msk - 1);
                int i   = ((p8 ^ low) << 1) | low;
                int j   = i | msk;
                float2 si = st[i], sj = st[j];
                st[i] = cadd(cmul(m00, si), cmul(m01, sj));
                st[j] = cadd(cmul(m01, si), cmul(m00, sj));
            }
        }
    }

    float z[4];
    #pragma unroll
    for (int w = 0; w < 4; ++w) {
        int msk = 1 << (3 - w);
        float a = 0.f;
        #pragma unroll
        for (int i = 0; i < 16; ++i) {
            float p = st[i].x * st[i].x + st[i].y * st[i].y;
            a += (i & msk) ? -p : p;
        }
        z[w] = a;
    }

    float* o = out + (size_t)b * 10;
    #pragma unroll
    for (int q = 0; q < 10; ++q) {
        float a = cb[q];
        #pragma unroll
        for (int j = 0; j < 4; ++j) a += cw[q * 4 + j] * z[j];
        o[q] = a;
    }
}

// ---------------------------------------------------------------------------
extern "C" void kernel_launch(void* const* d_in, const int* in_sizes, int n_in,
                              void* d_out, int out_size, void* d_ws, size_t ws_size,
                              hipStream_t stream)
{
    const float* x   = (const float*)d_in[0];
    const float* w1  = (const float*)d_in[1];
    const float* b1  = (const float*)d_in[2];
    const float* w2  = (const float*)d_in[3];
    const float* b2  = (const float*)d_in[4];
    const float* fw1 = (const float*)d_in[5];
    const float* fb1 = (const float*)d_in[6];
    const float* fw2 = (const float*)d_in[7];
    const float* fb2 = (const float*)d_in[8];
    const float* qw  = (const float*)d_in[9];
    const float* cw  = (const float*)d_in[10];
    const float* cb  = (const float*)d_in[11];
    float* out = (float*)d_out;

    const int B = in_sizes[0] / 784;          // 16384

    _Float16* h2 = (_Float16*)d_ws;                                 // B*256 f16
    _Float16* h3 = (_Float16*)((char*)d_ws + (size_t)B * 256 * 2);  // B*128 f16

    conv_fused_kernel<<<B, 128, 0, stream>>>(x, w1, b1, w2, b2, h2);
    fc1_kernel<<<B / 64, 256, 0, stream>>>(h2, fw1, fb1, h3);
    quantum_kernel<<<(B + 255) / 256, 256, 0, stream>>>(h3, fw2, fb2, qw, cw, cb, out, B);
}
